// BD_65463891525764
// MI455X (gfx1250) — compile-verified, matrix operand order checked by reading, not run on hardware
//
#include <hip/hip_runtime.h>

typedef _Float16 v16h __attribute__((ext_vector_type(16)));
typedef _Float16 v8h  __attribute__((ext_vector_type(8)));
typedef float    v8f  __attribute__((ext_vector_type(8)));

#define N_ROWS 262144
#define D_IN   80
#define D_OUT  160
#define KP     96          // padded K: 80 data + ones col (80) + zeros
#define EPSV   1e-5f

#define WMMA_F16(a, b, c) \
  __builtin_amdgcn_wmma_f32_16x16x32_f16(false, (a), false, (b), (short)0, (c), false, false)

// ------------------------------------------------------------------
// Load one 16x32 f16 WMMA A/B operand from a row-major [row][K] array.
// ISA layout (CDNA5 ISA): lane L holds row (L&15); elements 0..7 = K
// kbase+0..7, elements 8..15 = K kbase+16..23, kbase = (L<16)?0:8.
// Two contiguous 16-byte reads per lane (ds_read_b128 when from LDS).
// ------------------------------------------------------------------
__device__ __forceinline__ v16h load_ab(const _Float16* __restrict__ base,
                                        int stride_h, int row0, int kofs) {
  const int l = threadIdx.x & 31;
  const _Float16* p = base + (size_t)(row0 + (l & 15)) * stride_h + kofs +
                      ((l & 16) ? 8 : 0);
  v8h lo = *(const v8h*)(p);
  v8h hi = *(const v8h*)(p + 16);
  v16h r;
#pragma unroll
  for (int i = 0; i < 8; ++i) { r[i] = lo[i]; r[i + 8] = hi[i]; }
  return r;
}

// ==================================================================
// Kernel 1: S[mat] = Xpad^T * Xpad  (96x96, ones column at index 80)
// grid (64, 3), block 192 (6 waves). Wave w owns tile-row mi=w of the
// 6x6 output grid: its A operands (4 K-steps) are cached in registers
// for the whole 128-row chunk; only B tiles stream from LDS.
// ==================================================================
__global__ void BD_stats_kernel(const float* __restrict__ sub,
                                const float* __restrict__ left,
                                const float* __restrict__ right,
                                float* __restrict__ S) {
  const int mat = blockIdx.y;
  const float* __restrict__ X = (mat == 0) ? sub : (mat == 1) ? left : right;
  float* __restrict__ Sm = S + (size_t)mat * (KP * KP);

  // [channel][row-in-chunk], 128 rows, stride 136 halfwords (272B:
  // 16B aligned, bank step 4 mod 64 -> conflict-free b128 reads)
  __shared__ _Float16 XT[KP * 136];

  const int tid = threadIdx.x;          // 192 threads
  const int w = tid >> 5;               // wave id 0..5 == mi

  // pad channels 80..95: ones column at 80, zeros elsewhere (set once)
  for (int idx = tid; idx < 16 * 128; idx += 192) {
    int c = 80 + (idx >> 7);
    int r = idx & 127;
    XT[c * 136 + r] = (c == 80) ? (_Float16)1.0f : (_Float16)0.0f;
  }

  v8f acc[6];
#pragma unroll
  for (int t = 0; t < 6; ++t)
#pragma unroll
    for (int g = 0; g < 8; ++g) acc[t][g] = 0.0f;

  const int rows_per_block = N_ROWS / gridDim.x;   // 4096
  const int row_base = blockIdx.x * rows_per_block;

  for (int chunk = 0; chunk < rows_per_block / 128; ++chunk) {
    const int r0 = row_base + chunk * 128;
    __syncthreads();
    for (int idx = tid; idx < 128 * D_IN; idx += 192) {
      int r = idx / D_IN;
      int c = idx - r * D_IN;
      XT[c * 136 + r] = (_Float16)X[(size_t)(r0 + r) * D_IN + c];
    }
    __syncthreads();

    // cache this wave's A operands (A[m][k] = XT[mi*16+m][k])
    v16h a[4];
#pragma unroll
    for (int kc = 0; kc < 4; ++kc)
      a[kc] = load_ab(XT, 136, w * 16, kc * 32);

#pragma unroll
    for (int ni = 0; ni < 6; ++ni) {
#pragma unroll
      for (int kc = 0; kc < 4; ++kc) {   // 128 rows = 4 x K32
        v16h b = load_ab(XT, 136, ni * 16, kc * 32);  // B col n = XT row
        acc[ni] = WMMA_F16(a[kc], b, acc[ni]);
      }
    }
  }

  const int l = tid & 31;
  const int half = l >> 4;
  const int ncol = l & 15;
#pragma unroll
  for (int ni = 0; ni < 6; ++ni) {
#pragma unroll
    for (int g = 0; g < 8; ++g) {
      int row = w * 16 + g + 8 * half;   // C/D layout: VGPR g -> M=g(+8)
      int col = ni * 16 + ncol;
      atomicAdd(&Sm[row * KP + col], acc[ni][g]);
    }
  }
}

// ==================================================================
// Kernel 2: fold BatchNorm into f16 weights.
//   var_j = w_j^T E[xx^T] w_j - (w_j.xbar)^2   (bias cancels in BN)
//   W'[j][c<80] = alpha_j * W[j][c];  W'[j][80] = beta_j - alpha_j*(w_j.xbar)
// grid 3, block 256.
// ==================================================================
__global__ void BD_fold_kernel(const float* __restrict__ S,
    const float* __restrict__ W0, const float* __restrict__ b0,
    const float* __restrict__ g0, const float* __restrict__ be0,
    const float* __restrict__ W1, const float* __restrict__ b1,
    const float* __restrict__ g1, const float* __restrict__ be1,
    const float* __restrict__ W2, const float* __restrict__ b2,
    const float* __restrict__ g2, const float* __restrict__ be2,
    _Float16* __restrict__ Wp) {
  const int mat = blockIdx.x;
  const float* __restrict__ W  = mat == 0 ? W0  : mat == 1 ? W1  : W2;
  const float* __restrict__ gm = mat == 0 ? g0  : mat == 1 ? g1  : g2;
  const float* __restrict__ be = mat == 0 ? be0 : mat == 1 ? be1 : be2;
  (void)b0; (void)b1; (void)b2;
  const float* __restrict__ Sm = S + (size_t)mat * (KP * KP);
  _Float16* __restrict__ Wout = Wp + (size_t)mat * (D_OUT * KP);

  __shared__ float xbar[D_IN];
  __shared__ float wx[D_OUT];
  __shared__ float qsum[D_OUT];
  __shared__ float alpha[D_OUT];
  __shared__ float ct[D_OUT];

  const int tid = threadIdx.x;
  const float invN = 1.0f / (float)N_ROWS;

  if (tid < D_IN)  xbar[tid] = Sm[80 * KP + tid] * invN;  // colsum/N
  if (tid < D_OUT) qsum[tid] = 0.0f;
  __syncthreads();

  if (tid < D_OUT) {
    float s = 0.0f;
    for (int c = 0; c < D_IN; ++c) s += W[tid * D_IN + c] * xbar[c];
    wx[tid] = s;
  }
  __syncthreads();

  // quadratic form, parallel over (j, c) tasks
  for (int task = tid; task < D_OUT * D_IN; task += blockDim.x) {
    int j = task / D_IN;
    int c = task - j * D_IN;
    const float* __restrict__ Sr = Sm + c * KP;
    const float* __restrict__ Wj = W + j * D_IN;
    float u = 0.0f;
    for (int d = 0; d < D_IN; ++d) u += Wj[d] * Sr[d];
    atomicAdd(&qsum[j], Wj[c] * u);
  }
  __syncthreads();

  if (tid < D_OUT) {
    float q = qsum[tid] * invN;              // E[(w.x)^2]
    float var = q - wx[tid] * wx[tid];       // biased variance of h
    float a = gm[tid] * rsqrtf(var + EPSV);
    alpha[tid] = a;
    ct[tid] = be[tid] - a * wx[tid];
  }
  __syncthreads();

  for (int idx = tid; idx < D_OUT * KP; idx += blockDim.x) {
    int j = idx / KP;
    int c = idx - j * KP;
    float v;
    if (c < D_IN)     v = alpha[j] * W[j * D_IN + c];
    else if (c == 80) v = ct[j];
    else              v = 0.0f;
    Wout[idx] = (_Float16)v;
  }
}

// ==================================================================
// Kernel 3: per 64-row tile, recompute R_sub/R_left/R_right via WMMA,
// rowwise dots -> softmax -> combine original fp32 inputs.
// grid 4096, block 128 (4 waves, one 16-row M-tile each).
// ==================================================================
__global__ void BD_main_kernel(const float* __restrict__ sub,
                               const float* __restrict__ left,
                               const float* __restrict__ right,
                               const _Float16* __restrict__ Wp,
                               float* __restrict__ out) {
  __shared__ _Float16 xt[3][64 * 104];   // [row][chan], stride 104 (208B)
  __shared__ float pw[64][4];

  const int tid = threadIdx.x;
  const int rows0 = blockIdx.x * 64;

#pragma unroll
  for (int mat = 0; mat < 3; ++mat) {
    const float* __restrict__ X = mat == 0 ? sub : mat == 1 ? left : right;
    for (int idx = tid; idx < 64 * D_IN; idx += 128) {
      int r = idx / D_IN;
      int c = idx - r * D_IN;
      xt[mat][r * 104 + c] = (_Float16)X[(size_t)(rows0 + r) * D_IN + c];
    }
    for (int idx = tid; idx < 64 * 16; idx += 128) {
      int r = idx >> 4;
      int c = 80 + (idx & 15);
      xt[mat][r * 104 + c] = (c == 80) ? (_Float16)1.0f : (_Float16)0.0f;
    }
  }
  __syncthreads();

  const int w = tid >> 5;                 // M-tile: rows w*16 .. w*16+15

  // Hoist A operands: 3 matrices x 3 K-chunks (72 VGPRs)
  v16h A[3][3];
#pragma unroll
  for (int m = 0; m < 3; ++m)
#pragma unroll
    for (int kc = 0; kc < 3; ++kc)
      A[m][kc] = load_ab(xt[m], 104, w * 16, kc * 32);

  float dSL[8], dSR[8], dLR[8];
#pragma unroll
  for (int g = 0; g < 8; ++g) { dSL[g] = 0.f; dSR[g] = 0.f; dLR[g] = 0.f; }

#pragma unroll 2
  for (int nt = 0; nt < 10; ++nt) {       // 160 output channels
    v8f rs, rl, rr;
#pragma unroll
    for (int g = 0; g < 8; ++g) { rs[g] = 0.f; rl[g] = 0.f; rr[g] = 0.f; }
#pragma unroll
    for (int kc = 0; kc < 3; ++kc) {      // K = 96
      v16h bs = load_ab(Wp + 0 * D_OUT * KP, KP, nt * 16, kc * 32);
      rs = WMMA_F16(A[0][kc], bs, rs);
      v16h bl = load_ab(Wp + 1 * D_OUT * KP, KP, nt * 16, kc * 32);
      rl = WMMA_F16(A[1][kc], bl, rl);
      v16h br = load_ab(Wp + 2 * D_OUT * KP, KP, nt * 16, kc * 32);
      rr = WMMA_F16(A[2][kc], br, rr);
    }
#pragma unroll
    for (int g = 0; g < 8; ++g) {         // per-lane dot partials
      dSL[g] += rs[g] * rl[g];
      dSR[g] += rs[g] * rr[g];
      dLR[g] += rl[g] * rr[g];
    }
  }

  // reduce over the 16 lanes sharing each row (butterfly within halves)
#pragma unroll
  for (int g = 0; g < 8; ++g)
#pragma unroll
    for (int m = 1; m < 16; m <<= 1) {
      dSL[g] += __shfl_xor(dSL[g], m, 32);
      dSR[g] += __shfl_xor(dSR[g], m, 32);
      dLR[g] += __shfl_xor(dLR[g], m, 32);
    }

  const int l = tid & 31;
  if ((l & 15) == 0) {
    const int half = l >> 4;
#pragma unroll
    for (int g = 0; g < 8; ++g) {
      int r = w * 16 + g + 8 * half;
      float sl = dSL[g], sr = dSR[g], lr = dLR[g];
      float mx = fmaxf(sl, fmaxf(sr, lr));
      float e0 = __expf(sl - mx);
      float e1 = __expf(sr - mx);
      float e2 = __expf(lr - mx);
      float inv = 1.0f / (e0 + e1 + e2);
      pw[r][0] = e0 * inv;   // weight for left
      pw[r][1] = e1 * inv;   // weight for right
      pw[r][2] = e2 * inv;   // weight for sub
    }
  }
  __syncthreads();

  // combine original fp32 inputs (tiles are L1/L2-hot from the staging read)
  for (int idx = tid; idx < 64 * D_IN; idx += 128) {
    int r = idx / D_IN;
    int c = idx - r * D_IN;
    size_t go = (size_t)(rows0 + r) * D_IN + c;
    out[go] = pw[r][0] * left[go] + pw[r][1] * right[go] + pw[r][2] * sub[go];
  }
}

extern "C" void kernel_launch(void* const* d_in, const int* in_sizes, int n_in,
                              void* d_out, int out_size, void* d_ws, size_t ws_size,
                              hipStream_t stream) {
  (void)in_sizes; (void)n_in; (void)out_size; (void)ws_size;
  const float* sub   = (const float*)d_in[0];
  const float* left  = (const float*)d_in[1];
  const float* right = (const float*)d_in[2];
  const float* W_s  = (const float*)d_in[3];
  const float* b_s  = (const float*)d_in[4];
  const float* g_s  = (const float*)d_in[5];
  const float* be_s = (const float*)d_in[6];
  const float* W_l  = (const float*)d_in[7];
  const float* b_l  = (const float*)d_in[8];
  const float* g_l  = (const float*)d_in[9];
  const float* be_l = (const float*)d_in[10];
  const float* W_r  = (const float*)d_in[11];
  const float* b_r  = (const float*)d_in[12];
  const float* g_r  = (const float*)d_in[13];
  const float* be_r = (const float*)d_in[14];
  float* out = (float*)d_out;

  float* S = (float*)d_ws;                               // 3*96*96 f32
  _Float16* Wp = (_Float16*)((char*)d_ws + 3 * KP * KP * sizeof(float));

  hipMemsetAsync(S, 0, 3 * KP * KP * sizeof(float), stream);

  dim3 g1(64, 3, 1);
  BD_stats_kernel<<<g1, 192, 0, stream>>>(sub, left, right, S);

  BD_fold_kernel<<<3, 256, 0, stream>>>(S,
      W_s, b_s, g_s, be_s, W_l, b_l, g_l, be_l, W_r, b_r, g_r, be_r, Wp);

  BD_main_kernel<<<N_ROWS / 64, 128, 0, stream>>>(sub, left, right, Wp, out);
}
